// CIFAR100HyperViT_73375221284967
// MI455X (gfx1250) — matrix operand order repaired and checked by reading, not compiled, not run on hardware
//
#include <hip/hip_runtime.h>
#include <hip/hip_bf16.h>

#define TOK   8320      // 128 * 65
#define DMODEL 192

typedef __attribute__((ext_vector_type(16))) __bf16 v16bf;
typedef __attribute__((ext_vector_type(8)))  __bf16 v8bf;
typedef __attribute__((ext_vector_type(8)))  float  v8f;

union BFrag { v16bf v; v8bf h[2]; };

#define WMMA_BF16(a, b, c) \
  __builtin_amdgcn_wmma_f32_16x16x32_bf16(false, (a), false, (b), (short)0, (c), false, false)

// A fragment (16x32 bf16, row-major in LDS, stride in elements, 16B aligned).
// Lane 0-15: row=lane, K = {0..7, 16..23}; lane 16-31: row=lane-16, K = {8..15, 24..31}.
__device__ __forceinline__ v16bf afrag_lds(const __bf16* base, int stride) {
  int lane = threadIdx.x & 31;
  const __bf16* p = base + (lane & 15) * stride + ((lane >> 4) << 3);
  BFrag f;
  f.h[0] = *(const v8bf*)(p);
  f.h[1] = *(const v8bf*)(p + 16);
  return f.v;
}

// B fragment (32x16 bf16) from global weights stored (N,K) row-major:
// lane 0-15 holds column n=lane, K=kk+{0..7,16..23} ; lane 16-31: K=kk+{8..15,24..31}.
__device__ __forceinline__ v16bf bfrag_glb(const __bf16* W, int col0, int K, int kk) {
  int lane = threadIdx.x & 31;
  const __bf16* p = W + (size_t)(col0 + (lane & 15)) * K + kk + ((lane >> 4) << 3);
  BFrag f;
  f.h[0] = *(const v8bf*)(p);
  f.h[1] = *(const v8bf*)(p + 16);
  return f.v;
}

__device__ __forceinline__ float gelu_f(float x) {
  return 0.5f * x * (1.0f + erff(x * 0.70710678118654752440f));
}

// C tile epilogue: writes/accumulates one 16x16 f32 tile.
// C layout: lane 0-15 -> n=lane, M=v; lane 16-31 -> n=lane-16, M=8+v.
__device__ __forceinline__ void epi_tile(v8f acc, int row0, int col0, int N,
                                         const float* bias, float* out,
                                         float* resid, float sc) {
  int lane = threadIdx.x & 31;
  int col = col0 + (lane & 15);
  int rbase = row0 + ((lane >> 4) << 3);
  float bv = bias ? bias[col] : 0.0f;
  if (resid) {
    for (int v = 0; v < 8; ++v)
      resid[(size_t)(rbase + v) * N + col] += sc * (acc[v] + bv);
  } else {
    for (int v = 0; v < 8; ++v)
      out[(size_t)(rbase + v) * N + col] = acc[v] + bv;
  }
}

// Block reduction over first 192 lanes; safe for blockDim 192 or 384.
__device__ __forceinline__ float blk_reduce192(float v, float* red) {
  int tid = threadIdx.x;
  if (tid < 192) red[tid] = v;
  __syncthreads();
  if (tid < 96) red[tid] += red[tid + 96];
  __syncthreads();
  if (tid < 48) red[tid] += red[tid + 48];
  __syncthreads();
  if (tid < 24) red[tid] += red[tid + 24];
  __syncthreads();
  if (tid < 12) red[tid] += red[tid + 12];
  __syncthreads();
  if (tid < 6)  red[tid] += red[tid + 6];
  __syncthreads();
  float r = red[0] + red[1] + red[2] + red[3] + red[4] + red[5];
  __syncthreads();
  return r;
}

// ---------------- fp32 -> bf16 weight conversion ----------------
__global__ void f2bf(const float* __restrict__ s, __bf16* __restrict__ d, int n) {
  int i = blockIdx.x * 256 + threadIdx.x;
  if (i < n) d[i] = (__bf16)s[i];
}

// ---------------- patch embed + LN + pos embed ----------------
__global__ void patch_k(const float* __restrict__ x, const float* __restrict__ cw,
                        const float* __restrict__ cb, const float* __restrict__ peg,
                        const float* __restrict__ peb, const float* __restrict__ pos,
                        float* __restrict__ t) {
  __shared__ float red[192];
  int p = blockIdx.x, b = blockIdx.y, d = threadIdx.x;
  int pr = (p >> 3) * 4, pc = (p & 7) * 4;
  float acc = cb[d];
  for (int c = 0; c < 3; ++c)
    for (int i = 0; i < 4; ++i)
      for (int j = 0; j < 4; ++j)
        acc += x[(((size_t)b * 3 + c) * 32 + pr + i) * 32 + pc + j] *
               cw[d * 48 + c * 16 + i * 4 + j];
  float s = blk_reduce192(acc, red);
  float mean = s * (1.0f / 192.0f);
  float dv = acc - mean;
  float s2 = blk_reduce192(dv * dv, red);
  float var = s2 * (1.0f / 192.0f);
  float val = dv * rsqrtf(var + 1e-5f) * peg[d] + peb[d];
  t[((size_t)b * 65 + 1 + p) * DMODEL + d] = val + pos[(1 + p) * DMODEL + d];
}

__global__ void cls_k(const float* __restrict__ cls, const float* __restrict__ pos,
                      float* __restrict__ t) {
  int b = blockIdx.x, d = threadIdx.x;
  t[(size_t)b * 65 * DMODEL + d] = cls[d] + pos[d];
}

// ---------------- LayerNorm (rows of 192) ----------------
__global__ void ln_k(const float* __restrict__ in, float* __restrict__ out,
                     const float* __restrict__ g, const float* __restrict__ b) {
  __shared__ float red[192];
  int row = blockIdx.x, tid = threadIdx.x;
  float v = in[(size_t)row * DMODEL + tid];
  float s = blk_reduce192(v, red);
  float mean = s * (1.0f / 192.0f);
  float dv = v - mean;
  float s2 = blk_reduce192(dv * dv, red);
  float var = s2 * (1.0f / 192.0f);
  out[(size_t)row * DMODEL + tid] = dv * rsqrtf(var + 1e-5f) * g[tid] + b[tid];
}

// ---------------- WMMA GEMM: C[M,N] = A[M,192] @ W[N,192]^T (+bias) ----------------
// block = 256 threads = 8 waves (4 M x 2 N), wave tile 32x32 (4 accumulators),
// block tile 128x64.  Optional fused residual: resid += rscale * (C + bias).
__global__ __launch_bounds__(256) void gemm_k192(
    const float* __restrict__ A, const __bf16* __restrict__ W,
    const float* __restrict__ bias, float* __restrict__ out,
    float* __restrict__ resid, const float* __restrict__ rscale, int N) {
  __shared__ __align__(16) __bf16 As[128 * 200];
  int tid = threadIdx.x;
  int m0 = blockIdx.x * 128;
  // Pull this block's 64x192 bf16 weight slice toward the WGP while staging A.
  {
    const __bf16* wslice = W + (size_t)blockIdx.y * 64 * 192;
    if (tid < 192) __builtin_prefetch(wslice + (size_t)tid * 64, 0, 3);
  }
  for (int i = tid; i < 128 * 192; i += 256) {
    int r = i / 192, c = i - r * 192;
    As[r * 200 + c] = (__bf16)A[(size_t)(m0 + r) * 192 + c];
  }
  __syncthreads();
  int wv = tid >> 5;
  int mw = (wv >> 1) * 32;
  int nw = blockIdx.y * 64 + (wv & 1) * 32;
  v8f acc00 = {}, acc01 = {}, acc10 = {}, acc11 = {};
  for (int kk = 0; kk < 192; kk += 32) {
    v16bf a0 = afrag_lds(As + mw * 200 + kk, 200);
    v16bf a1 = afrag_lds(As + (mw + 16) * 200 + kk, 200);
    v16bf b0 = bfrag_glb(W, nw, 192, kk);
    v16bf b1 = bfrag_glb(W, nw + 16, 192, kk);
    acc00 = WMMA_BF16(a0, b0, acc00);
    acc01 = WMMA_BF16(a0, b1, acc01);
    acc10 = WMMA_BF16(a1, b0, acc10);
    acc11 = WMMA_BF16(a1, b1, acc11);
  }
  float sc = resid ? rscale[0] : 0.0f;
  epi_tile(acc00, m0 + mw,      nw,      N, bias, out, resid, sc);
  epi_tile(acc01, m0 + mw,      nw + 16, N, bias, out, resid, sc);
  epi_tile(acc10, m0 + mw + 16, nw,      N, bias, out, resid, sc);
  epi_tile(acc11, m0 + mw + 16, nw + 16, N, bias, out, resid, sc);
}

// ---------------- banded attention (window +-3), head dim 64 ----------------
__global__ void attn_k(const float* __restrict__ qkv, const float* __restrict__ temp,
                       float* __restrict__ o, int l) {
  __shared__ float sc[8];
  int n = blockIdx.x, h = blockIdx.y, b = blockIdx.z, tid = threadIdx.x;
  int row = b * 65 + n;
  const float* qp = qkv + (size_t)row * 576 + h * 64;
  int m0 = n - 3; if (m0 < 0) m0 = 0;
  int m1 = n + 3; if (m1 > 64) m1 = 64;
  int cnt = m1 - m0 + 1;
  float scl = 0.125f / temp[l * 3 + h];
  if (tid < cnt) {
    int m = m0 + tid;
    const float* kp = qkv + (size_t)(b * 65 + m) * 576 + 192 + h * 64;
    float s = 0.f;
    for (int d2 = 0; d2 < 64; ++d2) s += qp[d2] * kp[d2];
    sc[tid] = s * scl;
  }
  __syncthreads();
  if (tid == 0) {
    float mx = -1e30f;
    for (int i = 0; i < cnt; ++i) mx = fmaxf(mx, sc[i]);
    float sum = 0.f;
    for (int i = 0; i < cnt; ++i) { sc[i] = expf(sc[i] - mx); sum += sc[i]; }
    float inv = 1.f / sum;
    for (int i = 0; i < cnt; ++i) sc[i] *= inv;
  }
  __syncthreads();
  float od = 0.f;
  for (int i = 0; i < cnt; ++i) {
    const float* vp = qkv + (size_t)(b * 65 + m0 + i) * 576 + 384 + h * 64;
    od += sc[i] * vp[tid];
  }
  o[(size_t)row * DMODEL + h * 64 + tid] = od;
}

// ---------------- gate: softmax over 4 experts, top-2, renormalize ----------------
__global__ void gate_k(const float* __restrict__ y, const float* __restrict__ gwt,
                       const float* __restrict__ gb, float* __restrict__ w) {
  int n = blockIdx.x * 256 + threadIdx.x;
  if (n >= TOK) return;
  float lg[4];
  const float* yr = y + (size_t)n * DMODEL;
  for (int e = 0; e < 4; ++e) {
    float a = gb[e];
    const float* wr = gwt + e * DMODEL;
    for (int c = 0; c < DMODEL; ++c) a += yr[c] * wr[c];
    lg[e] = a;
  }
  float mx = fmaxf(fmaxf(lg[0], lg[1]), fmaxf(lg[2], lg[3]));
  float p[4], s = 0.f;
  for (int e = 0; e < 4; ++e) { p[e] = expf(lg[e] - mx); s += p[e]; }
  for (int e = 0; e < 4; ++e) p[e] /= s;
  int i1 = 0;
  for (int e = 1; e < 4; ++e) if (p[e] > p[i1]) i1 = e;
  int i2 = -1;
  for (int e = 0; e < 4; ++e) if (e != i1 && (i2 < 0 || p[e] > p[i2])) i2 = e;
  float den = p[i1] + p[i2];
  for (int e = 0; e < 4; ++e)
    w[(size_t)n * 4 + e] = (e == i1 || e == i2) ? p[e] / den : 0.0f;
}

// ---------------- fused MoE: hidden (16x1536 bf16 in LDS) then weighted combine ----------------
__global__ __launch_bounds__(256) void moe_k(
    const float* __restrict__ y, const __bf16* __restrict__ w1,
    const float* __restrict__ b1, const __bf16* __restrict__ w2,
    const float* __restrict__ b2, const float* __restrict__ gw,
    float* __restrict__ t, const float* __restrict__ mscale) {
  __shared__ __align__(16) __bf16 Ys[16 * 200];
  __shared__ __align__(16) __bf16 Hs[16 * 1544];
  __shared__ float Gw[16 * 4];
  int tid = threadIdx.x;
  int t0 = blockIdx.x * 16;
  for (int i = tid; i < 16 * 192; i += 256) {
    int r = i / 192, c = i - r * 192;
    Ys[r * 200 + c] = (__bf16)y[(size_t)(t0 + r) * 192 + c];
  }
  if (tid < 64) Gw[tid] = gw[(size_t)t0 * 4 + tid];
  __syncthreads();
  int wv = tid >> 5, lane = tid & 31;
  int colL = lane & 15;
  int rL = (lane >> 4) << 3;
  // phase 1: hidden = gelu(y @ e_w1^T + b1) * gate_weight -> bf16 LDS
  // Each wave: 16x32 tile (A-fragment reused across 2 B tiles).
  for (int tp = wv; tp < 48; tp += 8) {
    int e = tp / 12;
    int h0 = (tp - e * 12) * 32;
    const __bf16* We = w1 + (size_t)e * 384 * 192;
    v8f acc0 = {}, acc1 = {};
    for (int kk = 0; kk < 192; kk += 32) {
      v16bf a = afrag_lds(Ys + kk, 200);
      v16bf b0 = bfrag_glb(We, h0, 192, kk);
      v16bf b1 = bfrag_glb(We, h0 + 16, 192, kk);
      acc0 = WMMA_BF16(a, b0, acc0);
      acc1 = WMMA_BF16(a, b1, acc1);
    }
    for (int half = 0; half < 2; ++half) {
      v8f acc = half ? acc1 : acc0;
      int col = h0 + half * 16 + colL;
      float bv = b1[e * 384 + col];
      for (int v = 0; v < 8; ++v) {
        int m = rL + v;
        float g = gelu_f(acc[v] + bv);
        Hs[m * 1544 + e * 384 + col] = (__bf16)(g * Gw[m * 4 + e]);
      }
    }
  }
  __syncthreads();
  // phase 2: t += mlp_scale * ( sum_e hidden_e @ e_w2[e]^T + sum_e w_e*b2[e] )
  // 6 column-pair tiles of 16x32; waves 0..5 take one each.
  float ms = mscale[0];
  for (int tp = wv; tp < 6; tp += 8) {
    int c0 = tp * 32;
    v8f acc0 = {}, acc1 = {};
    for (int kk = 0; kk < 1536; kk += 32) {
      int e = kk / 384;
      const __bf16* We = w2 + (size_t)e * 192 * 384;
      int kl = kk - e * 384;
      v16bf a = afrag_lds(Hs + kk, 1544);
      v16bf b0 = bfrag_glb(We, c0, 384, kl);
      v16bf b1 = bfrag_glb(We, c0 + 16, 384, kl);
      acc0 = WMMA_BF16(a, b0, acc0);
      acc1 = WMMA_BF16(a, b1, acc1);
    }
    for (int half = 0; half < 2; ++half) {
      v8f acc = half ? acc1 : acc0;
      int col = c0 + half * 16 + colL;
      for (int v = 0; v < 8; ++v) {
        int m = rL + v;
        float bsum = 0.f;
        for (int e = 0; e < 4; ++e) bsum += Gw[m * 4 + e] * b2[e * 192 + col];
        t[(size_t)(t0 + m) * DMODEL + col] += ms * (acc[v] + bsum);
      }
    }
  }
}

// ---------------- final LN(cls) + circulant hyper-head + classifier ----------------
__global__ void head_k(const float* __restrict__ t, const float* __restrict__ ng,
                       const float* __restrict__ nb, const float* __restrict__ hcw,
                       const float* __restrict__ hcb, const float* __restrict__ hw,
                       const float* __restrict__ hb, float* __restrict__ out) {
  __shared__ float red[192];
  __shared__ float cls[192];
  __shared__ float hh[384];
  int b = blockIdx.x, tid = threadIdx.x;
  float v = (tid < 192) ? t[(size_t)b * 65 * DMODEL + tid] : 0.f;
  float s = blk_reduce192(v, red);
  float mean = s * (1.0f / 192.0f);
  float dv = v - mean;
  float s2 = blk_reduce192(dv * dv, red);
  float var = s2 * (1.0f / 192.0f);
  if (tid < 192) cls[tid] = dv * rsqrtf(var + 1e-5f) * ng[tid] + nb[tid];
  __syncthreads();
  {
    int k = tid / 96, o = tid - k * 96;
    float a = hcb[tid];
    for (int j = 0; j < 4; ++j) {
      int pidx = (k - j) & 3;                       // (k-j) mod 4
      const float* wrow = hcw + ((size_t)pidx * 96 + o) * 48;
      const float* cj = cls + j * 48;
      for (int c = 0; c < 48; ++c) a += cj[c] * wrow[c];
    }
    hh[tid] = gelu_f(a);
  }
  __syncthreads();
  if (tid < 100) {
    float a = hb[tid];
    const float* wr = hw + (size_t)tid * 384;
    for (int i = 0; i < 384; ++i) a += hh[i] * wr[i];
    out[(size_t)b * 100 + tid] = a;
  }
}

extern "C" void kernel_launch(void* const* d_in, const int* in_sizes, int n_in,
                              void* d_out, int out_size, void* d_ws, size_t ws_size,
                              hipStream_t stream) {
  const float* x       = (const float*)d_in[0];
  const float* conv_w  = (const float*)d_in[1];
  const float* conv_b  = (const float*)d_in[2];
  const float* pe_g    = (const float*)d_in[3];
  const float* pe_b    = (const float*)d_in[4];
  const float* cls_tok = (const float*)d_in[5];
  const float* pos_emb = (const float*)d_in[6];
  const float* n1_g    = (const float*)d_in[7];
  const float* n1_b    = (const float*)d_in[8];
  const float* qkv_w   = (const float*)d_in[9];
  const float* temp    = (const float*)d_in[10];
  const float* proj_w  = (const float*)d_in[11];
  const float* proj_b  = (const float*)d_in[12];
  const float* n2_g    = (const float*)d_in[13];
  const float* n2_b    = (const float*)d_in[14];
  const float* gate_w  = (const float*)d_in[15];
  const float* gate_b  = (const float*)d_in[16];
  const float* e_w1    = (const float*)d_in[17];
  const float* e_b1    = (const float*)d_in[18];
  const float* e_w2    = (const float*)d_in[19];
  const float* e_b2    = (const float*)d_in[20];
  const float* attn_sc = (const float*)d_in[21];
  const float* mlp_sc  = (const float*)d_in[22];
  const float* norm_g  = (const float*)d_in[23];
  const float* norm_b  = (const float*)d_in[24];
  const float* hc_w    = (const float*)d_in[25];
  const float* hc_b    = (const float*)d_in[26];
  const float* head_w  = (const float*)d_in[27];
  const float* head_b  = (const float*)d_in[28];
  float* out = (float*)d_out;

  char* ws = (char*)d_ws;
  size_t off = 0;
  auto alloc = [&](size_t bytes) -> void* {
    void* p = ws + off;
    off += (bytes + 255) & ~(size_t)255;
    return p;
  };
  float* t     = (float*)alloc((size_t)TOK * DMODEL * 4);
  float* y     = (float*)alloc((size_t)TOK * DMODEL * 4);
  float* qkv   = (float*)alloc((size_t)TOK * 576 * 4);
  float* ob    = (float*)alloc((size_t)TOK * DMODEL * 4);
  float* gwb   = (float*)alloc((size_t)TOK * 4 * 4);
  __bf16* qkvw_bf  = (__bf16*)alloc((size_t)12 * 576 * 192 * 2);
  __bf16* projw_bf = (__bf16*)alloc((size_t)12 * 192 * 192 * 2);
  __bf16* ew1_bf   = (__bf16*)alloc((size_t)12 * 4 * 384 * 192 * 2);
  __bf16* ew2_bf   = (__bf16*)alloc((size_t)12 * 4 * 192 * 384 * 2);

  auto cv = [&](const float* s, __bf16* d, int n) {
    f2bf<<<(n + 255) / 256, 256, 0, stream>>>(s, d, n);
  };
  cv(qkv_w,  qkvw_bf,  12 * 576 * 192);
  cv(proj_w, projw_bf, 12 * 192 * 192);
  cv(e_w1,   ew1_bf,   12 * 4 * 384 * 192);
  cv(e_w2,   ew2_bf,   12 * 4 * 192 * 384);

  patch_k<<<dim3(64, 128), 192, 0, stream>>>(x, conv_w, conv_b, pe_g, pe_b, pos_emb, t);
  cls_k<<<128, 192, 0, stream>>>(cls_tok, pos_emb, t);

  for (int l = 0; l < 12; ++l) {
    ln_k<<<TOK, 192, 0, stream>>>(t, y, n1_g + l * 192, n1_b + l * 192);
    gemm_k192<<<dim3(65, 9), 256, 0, stream>>>(
        y, qkvw_bf + (size_t)l * 576 * 192, nullptr, qkv, nullptr, nullptr, 576);
    attn_k<<<dim3(65, 3, 128), 64, 0, stream>>>(qkv, temp, ob, l);
    gemm_k192<<<dim3(65, 3), 256, 0, stream>>>(
        ob, projw_bf + (size_t)l * 192 * 192, proj_b + l * 192, t, t, attn_sc + l, 192);
    ln_k<<<TOK, 192, 0, stream>>>(t, y, n2_g + l * 192, n2_b + l * 192);
    gate_k<<<(TOK + 255) / 256, 256, 0, stream>>>(y, gate_w + l * 4 * 192, gate_b + l * 4, gwb);
    moe_k<<<TOK / 16, 256, 0, stream>>>(
        y, ew1_bf + (size_t)l * 4 * 384 * 192, e_b1 + l * 4 * 384,
        ew2_bf + (size_t)l * 4 * 192 * 384, e_b2 + l * 4 * 192, gwb, t, mlp_sc + l);
  }

  head_k<<<128, 384, 0, stream>>>(t, norm_g, norm_b, hc_w, hc_b, head_w, head_b, out);
}
